// TensorBase_14826227106527
// MI455X (gfx1250) — compile-verified, matrix operand order checked by reading, not run on hardware
//
#include <hip/hip_runtime.h>
#include <hip/hip_bf16.h>

typedef __attribute__((ext_vector_type(16))) _Float16 v16h;
typedef __attribute__((ext_vector_type(8)))  float    v8f;

#define NRAYS 4096
#define NSAMP 256
#define FDIM  27
#define CH    128
#define KIN   66        // 27 + 3 + 36
#define K1PAD 96        // KIN padded to 3 * 32
#define KS1   3
#define KS2   4
#define NT    8         // 128 / 16 column tiles
#define FRAG  512       // halves per 16x32 fragment (32 lanes * 16 halves)
#define TAILP 40        // padded per-ray tail (39 used)
#define WBUF_HALVES 2176 // per-wave LDS scratch: max(3*512, 4*512, 16*136)

// workspace layout (in _Float16 units)
#define W1P_OFF 0
#define W1P_HALVES (KS1*NT*FRAG)               // 12288
#define W2P_OFF (W1P_OFF + W1P_HALVES)
#define W2P_HALVES (KS2*NT*FRAG)               // 16384
#define TAIL_OFF (W2P_OFF + W2P_HALVES)        // 28672
#define TAIL_HALVES (NRAYS*TAILP)              // 163840
#define WTS_OFF_H (TAIL_OFF + TAIL_HALVES)     // halves offset of float weights

#define DENSITY_SHIFT  (-10.0f)
#define DISTANCE_SCALE (25.0f)

__device__ __forceinline__ float sigmoidf_(float x) {
    return 1.0f / (1.0f + expf(-x));
}

// -------------------------------------------------------------------------
// Kernel 1: pack W1 (K padded to 96) and W2 (K=128) into the CDNA5 WMMA
// B-fragment layout (16x16x32 f16): fragment (ks, j) occupies 512 halves;
// lane L holds column n = j*16 + (L&15); v16h element h maps to
//   K = ks*32 + ((L>>4)<<3) + (h < 8 ? h : h + 8)
// so each lane's fragment is 32 contiguous bytes in LDS.
// -------------------------------------------------------------------------
__global__ void pack_weights_kernel(const float* __restrict__ W1,
                                    const float* __restrict__ W2,
                                    _Float16* __restrict__ W1p,
                                    _Float16* __restrict__ W2p) {
    int tid = blockIdx.x * blockDim.x + threadIdx.x;
    if (tid < W1P_HALVES) {
        int fid = tid >> 9;          // fragment id: ks*8 + j
        int within = tid & 511;
        int L = within >> 4, h = within & 15;
        int ks = fid >> 3, j = fid & 7;
        int n = j * 16 + (L & 15);
        int k = ks * 32 + ((L >> 4) << 3) + (h < 8 ? h : h + 8);
        float v = (k < KIN) ? W1[k * CH + n] : 0.0f;
        W1p[tid] = (_Float16)v;
    } else if (tid < W1P_HALVES + W2P_HALVES) {
        int t = tid - W1P_HALVES;
        int fid = t >> 9;
        int within = t & 511;
        int L = within >> 4, h = within & 15;
        int ks = fid >> 3, j = fid & 7;
        int n = j * 16 + (L & 15);
        int k = ks * 32 + ((L >> 4) << 3) + (h < 8 ? h : h + 8);
        W2p[t] = (_Float16)W2[k * CH + n];
    }
}

// -------------------------------------------------------------------------
// Kernel 2: per-ray work.
//  - tail[r][0..38] = [viewdirs(3), sin(pe)(18), cos(pe)(18)]  (f16)
//  - weights[r][s]  = alpha * T   (sequential cumprod, f32)
// -------------------------------------------------------------------------
__global__ void ray_prep_kernel(const float* __restrict__ density_feat,
                                const float* __restrict__ dists,
                                const float* __restrict__ viewdirs,
                                _Float16* __restrict__ tail,
                                float* __restrict__ wts) {
    int r = blockIdx.x * blockDim.x + threadIdx.x;
    if (r >= NRAYS) return;

    float v[3];
    v[0] = viewdirs[r * 3 + 0];
    v[1] = viewdirs[r * 3 + 1];
    v[2] = viewdirs[r * 3 + 2];
    _Float16* t = tail + (size_t)r * TAILP;
    t[0] = (_Float16)v[0];
    t[1] = (_Float16)v[1];
    t[2] = (_Float16)v[2];
    // pe index ordering from jnp reshape: idx = d*VIEW_PE + f
    for (int d = 0; d < 3; ++d) {
        float fb = 1.0f;
        for (int f = 0; f < 6; ++f) {
            float p = v[d] * fb;
            t[3 + d * 6 + f]      = (_Float16)sinf(p);
            t[3 + 18 + d * 6 + f] = (_Float16)cosf(p);
            fb *= 2.0f;
        }
    }

    const float* df = density_feat + (size_t)r * NSAMP;
    const float* dd = dists + (size_t)r * NSAMP;
    float* w = wts + (size_t)r * NSAMP;
    float T = 1.0f;
    for (int s = 0; s < NSAMP; ++s) {
        float x = df[s] + DENSITY_SHIFT;
        // numerically stable softplus
        float sp = (x > 0.0f) ? (x + log1pf(expf(-x))) : log1pf(expf(x));
        float alpha = 1.0f - expf(-sp * dd[s] * DISTANCE_SCALE);
        w[s] = alpha * T;
        T *= (1.0f - alpha + 1e-10f);
    }
}

// -------------------------------------------------------------------------
// Kernel 3: fused MLP + composite. One workgroup (8 waves) per ray.
// Each wave processes two 16-sample tiles (16 tiles per ray).
// -------------------------------------------------------------------------
__global__ __launch_bounds__(256, 1)
void render_mlp_kernel(const float* __restrict__ features,
                       const float* __restrict__ b1g,
                       const float* __restrict__ b2g,
                       const float* __restrict__ W3g,
                       const float* __restrict__ b3g,
                       const _Float16* __restrict__ W1p_g,
                       const _Float16* __restrict__ W2p_g,
                       const _Float16* __restrict__ tail_g,
                       const float* __restrict__ wts_g,
                       float* __restrict__ out) {
    extern __shared__ __attribute__((aligned(32))) char smem[];
    _Float16* sW1   = (_Float16*)smem;                        // 12288 halves
    _Float16* sW2   = sW1 + W1P_HALVES;                       // 16384 halves
    _Float16* sWave = sW2 + W2P_HALVES;                       // 8 * 2176 halves
    float* sW3 = (float*)(sWave + 8 * WBUF_HALVES);           // 384
    float* sB1 = sW3 + CH * 3;                                // 128
    float* sB2 = sB1 + CH;                                    // 128
    float* sB3 = sB2 + CH;                                    // 4
    float* sRed = sB3 + 4;                                    // 8 * 4

    const int tid  = threadIdx.x;
    const int wave = tid >> 5;
    const int lane = tid & 31;
    const int r    = blockIdx.x;

    // cooperative stage of packed weights (W1p & W2p are contiguous in ws)
    {
        const uint4* g = (const uint4*)W1p_g;
        uint4* s = (uint4*)sW1;
        for (int i = tid; i < (W1P_HALVES + W2P_HALVES) / 8; i += 256) s[i] = g[i];
        for (int i = tid; i < CH * 3; i += 256) sW3[i] = W3g[i];
        for (int i = tid; i < CH; i += 256) { sB1[i] = b1g[i]; sB2[i] = b2g[i]; }
        if (tid < 3) sB3[tid] = b3g[tid];
    }
    __syncthreads();

    _Float16* myBuf = sWave + wave * WBUF_HALVES;
    const _Float16* tl = tail_g + (size_t)r * TAILP;

    float accw = 0.0f, a0 = 0.0f, a1 = 0.0f, a2 = 0.0f;
    const float lmask = (lane < 16) ? 1.0f : 0.0f;

    for (int t = 0; t < 2; ++t) {
        const int s0 = (wave + t * 8) * 16;

        // ---- stage A tile (16 rows x 96 cols) into fragment-packed layout ----
        // element (row, k) -> offset (k>>5)*512 + (row + 16*((k>>3)&1))*16
        //                     + (k&7) + 8*((k&31)>=16)
        const float* fbase = features + ((size_t)r * NSAMP + s0) * FDIM;
        for (int i = lane; i < 16 * FDIM; i += 32) {
            int row = i / FDIM, k = i % FDIM;
            int off = ((k >> 5) * FRAG) + (row + (((k >> 3) & 1) << 4)) * 16 +
                      (k & 7) + (((k & 31) >= 16) ? 8 : 0);
            myBuf[off] = (_Float16)fbase[i];
        }
        for (int i = lane; i < 16 * (K1PAD - FDIM); i += 32) {
            int row = i / (K1PAD - FDIM);
            int k = FDIM + i % (K1PAD - FDIM);
            _Float16 v = (k < KIN) ? tl[k - FDIM] : (_Float16)0.0f;
            int off = ((k >> 5) * FRAG) + (row + (((k >> 3) & 1) << 4)) * 16 +
                      (k & 7) + (((k & 31) >= 16) ? 8 : 0);
            myBuf[off] = v;
        }
        __builtin_amdgcn_wave_barrier();  // LDS per-wave: DS ops are in-order

        // ---- layer 1: [16x96] @ [96x128], f16 WMMA, f32 accumulate ----
        v8f acc[NT];
#pragma unroll
        for (int j = 0; j < NT; ++j) {
            float bv = sB1[j * 16 + (lane & 15)];
            v8f c;
#pragma unroll
            for (int q = 0; q < 8; ++q) c[q] = bv;
            acc[j] = c;
        }
#pragma unroll
        for (int ks = 0; ks < KS1; ++ks) {
            v16h a = *(const v16h*)(myBuf + ks * FRAG + lane * 16);
#pragma unroll
            for (int j = 0; j < NT; ++j) {
                v16h b = *(const v16h*)(sW1 + (ks * NT + j) * FRAG + lane * 16);
                acc[j] = __builtin_amdgcn_wmma_f32_16x16x32_f16(
                    false, a, false, b, (short)0, acc[j], false, false);
            }
        }
        __builtin_amdgcn_wave_barrier();

        // ---- relu + repack h1 into A-fragment layout for layer 2 ----
        // accumulator element (j, q, lane) = h1[row=q+8*(lane>>4)][k=16j+(lane&15)]
#pragma unroll
        for (int j = 0; j < NT; ++j) {
            int k = j * 16 + (lane & 15);
            int dhi = ((k >> 3) & 1) << 4;
            int hofs = (k & 7) + (((k & 31) >= 16) ? 8 : 0);
            _Float16* dst = myBuf + (k >> 5) * FRAG + hofs;
            int rbase = (lane >> 4) << 3;
#pragma unroll
            for (int q = 0; q < 8; ++q) {
                float x = acc[j][q];
                x = x > 0.0f ? x : 0.0f;
                dst[(q + rbase + dhi) * 16] = (_Float16)x;
            }
        }
        __builtin_amdgcn_wave_barrier();

        // ---- layer 2: [16x128] @ [128x128] ----
        v8f acc2[NT];
#pragma unroll
        for (int j = 0; j < NT; ++j) {
            float bv = sB2[j * 16 + (lane & 15)];
            v8f c;
#pragma unroll
            for (int q = 0; q < 8; ++q) c[q] = bv;
            acc2[j] = c;
        }
#pragma unroll
        for (int ks = 0; ks < KS2; ++ks) {
            v16h a = *(const v16h*)(myBuf + ks * FRAG + lane * 16);
#pragma unroll
            for (int j = 0; j < NT; ++j) {
                v16h b = *(const v16h*)(sW2 + (ks * NT + j) * FRAG + lane * 16);
                acc2[j] = __builtin_amdgcn_wmma_f32_16x16x32_f16(
                    false, a, false, b, (short)0, acc2[j], false, false);
            }
        }
        __builtin_amdgcn_wave_barrier();

        // ---- relu + store h2 row-major [16][136] for layer-3 dot ----
#pragma unroll
        for (int j = 0; j < NT; ++j) {
            int k = j * 16 + (lane & 15);
            int rbase = (lane >> 4) << 3;
#pragma unroll
            for (int q = 0; q < 8; ++q) {
                float x = acc2[j][q];
                x = x > 0.0f ? x : 0.0f;
                myBuf[(q + rbase) * 136 + k] = (_Float16)x;
            }
        }
        __builtin_amdgcn_wave_barrier();

        // ---- layer 3: 128 -> 3, split K across lane halves ----
        int row3 = lane & 15;
        int kb = (lane >> 4) << 6;  // 0 or 64
        float d0 = 0.0f, d1 = 0.0f, d2 = 0.0f;
#pragma unroll 4
        for (int kk = 0; kk < 64; ++kk) {
            float hv = (float)myBuf[row3 * 136 + kb + kk];
            const float* wp = sW3 + (kb + kk) * 3;
            d0 = fmaf(hv, wp[0], d0);
            d1 = fmaf(hv, wp[1], d1);
            d2 = fmaf(hv, wp[2], d2);
        }
        d0 += __shfl_xor(d0, 16, 32);
        d1 += __shfl_xor(d1, 16, 32);
        d2 += __shfl_xor(d2, 16, 32);

        float r0 = sigmoidf_(d0 + sB3[0]);
        float r1 = sigmoidf_(d1 + sB3[1]);
        float r2 = sigmoidf_(d2 + sB3[2]);
        float w = wts_g[(size_t)r * NSAMP + s0 + row3] * lmask;
        accw += w;
        a0 += w * r0;
        a1 += w * r1;
        a2 += w * r2;
    }

    // reduce across the 32 lanes of this wave
    for (int m = 16; m >= 1; m >>= 1) {
        accw += __shfl_xor(accw, m, 32);
        a0   += __shfl_xor(a0,   m, 32);
        a1   += __shfl_xor(a1,   m, 32);
        a2   += __shfl_xor(a2,   m, 32);
    }
    if (lane == 0) {
        sRed[wave * 4 + 0] = accw;
        sRed[wave * 4 + 1] = a0;
        sRed[wave * 4 + 2] = a1;
        sRed[wave * 4 + 3] = a2;
    }
    __syncthreads();
    if (tid == 0) {
        float tw = 0, t0 = 0, t1 = 0, t2 = 0;
        for (int wv = 0; wv < 8; ++wv) {
            tw += sRed[wv * 4 + 0];
            t0 += sRed[wv * 4 + 1];
            t1 += sRed[wv * 4 + 2];
            t2 += sRed[wv * 4 + 3];
        }
        float bg = 1.0f - tw;  // white background
        out[(size_t)r * 3 + 0] = t0 + bg;
        out[(size_t)r * 3 + 1] = t1 + bg;
        out[(size_t)r * 3 + 2] = t2 + bg;
    }
}

extern "C" void kernel_launch(void* const* d_in, const int* in_sizes, int n_in,
                              void* d_out, int out_size, void* d_ws, size_t ws_size,
                              hipStream_t stream) {
    const float* density_feat = (const float*)d_in[0];
    const float* dists        = (const float*)d_in[1];
    const float* features     = (const float*)d_in[2];
    const float* viewdirs     = (const float*)d_in[3];
    const float* W1 = (const float*)d_in[4];
    const float* b1 = (const float*)d_in[5];
    const float* W2 = (const float*)d_in[6];
    const float* b2 = (const float*)d_in[7];
    const float* W3 = (const float*)d_in[8];
    const float* b3 = (const float*)d_in[9];
    float* out = (float*)d_out;

    _Float16* h = (_Float16*)d_ws;
    _Float16* W1p  = h + W1P_OFF;
    _Float16* W2p  = h + W2P_OFF;
    _Float16* tail = h + TAIL_OFF;
    float* wts = (float*)(h + WTS_OFF_H);

    {
        int n = W1P_HALVES + W2P_HALVES;  // 28672 packing threads
        pack_weights_kernel<<<(n + 255) / 256, 256, 0, stream>>>(W1, W2, W1p, W2p);
    }
    ray_prep_kernel<<<(NRAYS + 255) / 256, 256, 0, stream>>>(
        density_feat, dists, viewdirs, tail, wts);

    size_t smem = (size_t)(W1P_HALVES + W2P_HALVES + 8 * WBUF_HALVES) * sizeof(_Float16)
                + (size_t)(CH * 3 + CH + CH + 4 + 32) * sizeof(float);  // ~94.9 KB
    render_mlp_kernel<<<NRAYS, 256, smem, stream>>>(
        features, b1, b2, W3, b3, W1p, W2p, tail, wts, out);
}
